// MultiHeadAttention_41944650612760
// MI455X (gfx1250) — matrix operand order repaired and checked by reading, not compiled, not run on hardware
//
#include <hip/hip_runtime.h>

// ---------------- problem constants ----------------
#define BB   8
#define TT   1024
#define DD   512
#define HH   8
#define DKK  64
#define BT   8192   // BB*TT

typedef _Float16 half_t;
typedef __attribute__((ext_vector_type(16))) _Float16 v16h;
typedef __attribute__((ext_vector_type(8)))  _Float16 v8h;
typedef __attribute__((ext_vector_type(8)))  float    v8f;

// ---------------- WMMA helpers ----------------
__device__ inline v8f wmma_f16(v16h a, v16h b, v8f c) {
    return __builtin_amdgcn_wmma_f32_16x16x32_f16(false, a, false, b, (short)0, c, false, false);
}

__device__ inline v16h cat8(v8h a, v8h b) {
    return __builtin_shufflevector(a, b, 0,1,2,3,4,5,6,7,8,9,10,11,12,13,14,15);
}

// A 16x32 f16 layout: lane<16 holds row M=lane, k = {0..7, 16..23}; lane>=16: row M=lane-16, k = {8..15, 24..31}
__device__ inline v16h load_a16(const half_t* rowbase, int hi) {
    const half_t* p0 = rowbase + (hi ? 8 : 0);
    return cat8(*(const v8h*)p0, *(const v8h*)(p0 + 16));
}
// B 32x16 f16 layout: lane<16 holds col N=lane, k = {0..7, 16..23}; lane>=16: col N=lane-16, k = {8..15, 24..31}
__device__ inline v16h load_b16(const half_t* colbase, int hi) {
    const half_t* p = colbase + (hi ? 8 : 0);
    return cat8(*(const v8h*)p, *(const v8h*)(p + 16));
}
__device__ inline v16h load_a32(const float* rowbase, int hi) {
    const float* p0 = rowbase + (hi ? 8 : 0);
    const float* p1 = p0 + 16;
    v16h r;
#pragma unroll
    for (int i = 0; i < 8; ++i) { r[i] = (half_t)p0[i]; r[8 + i] = (half_t)p1[i]; }
    return r;
}
__device__ inline v16h load_b32(const float* colbase, int hi) {
    const float* p = colbase + (hi ? 8 : 0);
    v16h r;
#pragma unroll
    for (int i = 0; i < 8; ++i) { r[i] = (half_t)p[i]; r[8 + i] = (half_t)p[16 + i]; }
    return r;
}

// ---------------- wave32 reductions ----------------
__device__ inline float wave_max32(float v) {
#pragma unroll
    for (int off = 16; off > 0; off >>= 1) v = fmaxf(v, __shfl_xor(v, off, 32));
    return v;
}
__device__ inline float wave_sum32(float v) {
#pragma unroll
    for (int off = 16; off > 0; off >>= 1) v += __shfl_xor(v, off, 32);
    return v;
}
__device__ inline float wave_exscan32(float v) {
    int lane = threadIdx.x & 31;
    float x = v;
#pragma unroll
    for (int off = 1; off < 32; off <<= 1) {
        float y = __shfl_up(x, off, 32);
        if (lane >= off) x += y;
    }
    return x - v;
}

// =====================================================================
// Kernel 1: six projection GEMMs Y = X @ W^T + b  (64x64 tile per block)
//   bz: 0 km, 1 kc->sqrt, 2 qm, 3 qc->sqrt, 4 vm->transposed, 5 vc->transposed
// =====================================================================
__global__ __launch_bounds__(256) void proj_kernel(
    const float* q_mean, const float* q_cov, const float* k_mean, const float* k_cov,
    const float* v_mean, const float* v_cov,
    const float* Wk_mean, const float* bk_mean, const float* Wk_cov, const float* bk_cov,
    const float* Wv_mean, const float* bv_mean, const float* Wv_cov, const float* bv_cov,
    half_t* km_h, half_t* sqk_h, half_t* qm_h, half_t* sqq_h, half_t* vmT, half_t* vcT,
    float* kmsq, float* kcsum, float* qmsq, float* qcsum)
{
    __shared__ float Yl[64][65];

    const int bx = blockIdx.x, h = blockIdx.y, bz = blockIdx.z;
    const int m0 = bx * 64, n0 = h * 64;
    const int tid = threadIdx.x, lane = tid & 31, wv = tid >> 5;
    const int li = lane & 15, hi = (lane >= 16);

    const float *X, *W, *bias;
    switch (bz) {
        case 0:  X = k_mean; W = Wk_mean; bias = bk_mean; break;
        case 1:  X = k_cov;  W = Wk_cov;  bias = bk_cov;  break;
        case 2:  X = q_mean; W = Wk_mean; bias = bk_mean; break;
        case 3:  X = q_cov;  W = Wk_cov;  bias = bk_cov;  break;
        case 4:  X = v_mean; W = Wv_mean; bias = bv_mean; break;
        default: X = v_cov;  W = Wv_cov;  bias = bv_cov;  break;
    }

    // two 16x16 tiles per wave: tiles wv and wv+8 in a 4x4 tile grid
    const int t0i = wv, t1i = wv + 8;
    const int tr0 = t0i >> 2, tc0 = t0i & 3, tr1 = t1i >> 2, tc1 = t1i & 3;
    const float* ar0 = X + (size_t)(m0 + tr0 * 16 + li) * DD;
    const float* ar1 = X + (size_t)(m0 + tr1 * 16 + li) * DD;
    const float* bc0 = W + (size_t)(n0 + tc0 * 16 + li) * DD;
    const float* bc1 = W + (size_t)(n0 + tc1 * 16 + li) * DD;

    v8f acc0 = {}; v8f acc1 = {};
    for (int kb = 0; kb < DD; kb += 32) {
        acc0 = wmma_f16(load_a32(ar0 + kb, hi), load_b32(bc0 + kb, hi), acc0);
        acc1 = wmma_f16(load_a32(ar1 + kb, hi), load_b32(bc1 + kb, hi), acc1);
    }
    const float bn0 = bias[n0 + tc0 * 16 + li];
    const float bn1 = bias[n0 + tc1 * 16 + li];
#pragma unroll
    for (int i = 0; i < 8; ++i) {
        Yl[tr0 * 16 + i + 8 * hi][tc0 * 16 + li] = acc0[i] + bn0;
        Yl[tr1 * 16 + i + 8 * hi][tc1 * 16 + li] = acc1[i] + bn1;
    }
    __syncthreads();

    const int bb = m0 / TT;
    const int tbase = m0 % TT;

    if (bz < 4) {
        half_t* outh = (bz == 0) ? km_h : (bz == 1) ? sqk_h : (bz == 2) ? qm_h : sqq_h;
        const int r = tid >> 2, c0 = (tid & 3) * 16;
        half_t* dst = outh + ((size_t)bb * TT + tbase + r) * DD + n0 + c0;
        if (bz == 0 || bz == 2) {
#pragma unroll
            for (int c = 0; c < 16; ++c) dst[c] = (half_t)Yl[r][c0 + c];
        } else {
#pragma unroll
            for (int c = 0; c < 16; ++c) dst[c] = (half_t)sqrtf(fmaxf(Yl[r][c0 + c], 1e-24f));
        }
        if (tid < 64) {
            float s = 0.f;
            if (bz == 0 || bz == 2) {
                for (int c = 0; c < 64; ++c) { float y = Yl[tid][c]; s += y * y; }
            } else {
                for (int c = 0; c < 64; ++c) s += Yl[tid][c];
            }
            float* np = (bz == 0) ? kmsq : (bz == 1) ? kcsum : (bz == 2) ? qmsq : qcsum;
            np[(size_t)(bb * HH + h) * TT + tbase + tid] = s;
        }
    } else {
        half_t* outT = (bz == 4) ? vmT : vcT;
        const int r = tid >> 2, c0 = (tid & 3) * 16;
        const int t = tbase + r;
#pragma unroll
        for (int c = 0; c < 16; ++c) {
            int d = c0 + c;
            outT[((size_t)(bb * HH + h) * DKK + d) * TT + t] = (half_t)Yl[r][d];
        }
    }
}

// =====================================================================
// Kernel 2: attention. One block = 16 query rows of one (b,h).
// =====================================================================
union AttU {
    float sc[16][1024];                                  // raw scores, 64KB
    struct PH { half_t p[16][1024]; half_t q[16][1024]; } ph; // probs + probs^2, 64KB
};

__global__ __launch_bounds__(256) void attn_kernel(
    const half_t* qm_h, const half_t* sqq_h, const half_t* km_h, const half_t* sqk_h,
    const half_t* vmT, const half_t* vcT,
    const float* qmsq, const float* qcsum, const float* kmsq, const float* kcsum,
    half_t* cmh, half_t* cch, const float* gammas, const int* zero_pad)
{
    __shared__ AttU u;

    const int bt = blockIdx.x, by = blockIdx.y;   // by = b*H + h
    const int bb = by >> 3, h = by & 7;
    const int t0 = bt * 16;
    const int validN = t0 + 16;
    const int tid = threadIdx.x, lane = tid & 31, wv = tid >> 5;
    const int li = lane & 15, hi = (lane >= 16);

    // ---- hoisted A operands (qm rows, sqrt(qc) rows) ----
    const half_t* qbase = qm_h  + ((size_t)bb * TT + t0 + li) * DD + h * DKK;
    const half_t* sbase = sqq_h + ((size_t)bb * TT + t0 + li) * DD + h * DKK;
    v16h aq0 = load_a16(qbase, hi),      aq1 = load_a16(qbase + 32, hi);
    v16h as0 = load_a16(sbase, hi),      as1 = load_a16(sbase + 32, hi);

    float qn8[8];
#pragma unroll
    for (int i = 0; i < 8; ++i) {
        size_t idx = (size_t)by * TT + t0 + i + 8 * hi;
        qn8[i] = qmsq[idx] + qcsum[idx];
    }

    // ---- score pass: causal column tiles only ----
    const int ntiles = bt + 1;
    for (int st = wv; st < ntiles; st += 8) {
        const int s = st * 16 + li;
        const half_t* kbp = km_h  + ((size_t)bb * TT + s) * DD + h * DKK;
        const half_t* skp = sqk_h + ((size_t)bb * TT + s) * DD + h * DKK;
        v8f acc = {};
        acc = wmma_f16(aq0, load_b16(kbp, hi),      acc);
        acc = wmma_f16(aq1, load_b16(kbp + 32, hi), acc);
        acc = wmma_f16(as0, load_b16(skp, hi),      acc);
        acc = wmma_f16(as1, load_b16(skp + 32, hi), acc);
        const float kn = kmsq[(size_t)by * TT + s] + kcsum[(size_t)by * TT + s];
#pragma unroll
        for (int i = 0; i < 8; ++i) {
            int r = i + 8 * hi;
            u.sc[r][s] = (2.0f * acc[i] - qn8[i] - kn) * 0.125f;  // 1/sqrt(64)
        }
    }
    __syncthreads();

    // ---- row scan: load my segments into registers first ----
    const float g = gammas[h];
    const float gamma = -((g > 20.f) ? g : log1pf(expf(g)));   // -softplus
    const int zp = zero_pad[0];
    const int s0 = lane * 32;

    float raw[2][32];
#pragma unroll
    for (int rep = 0; rep < 2; ++rep) {
        const int r = wv * 2 + rep, t = t0 + r;
#pragma unroll
        for (int j = 0; j < 32; ++j) {
            int s = s0 + j;
            float v = u.sc[r][s];
            raw[rep][j] = (s <= t) ? v : -3.0e38f;
        }
    }
    __syncthreads();   // all reads of u.sc done; union reused as ph below

#pragma unroll
    for (int rep = 0; rep < 2; ++rep) {
        const int r = wv * 2 + rep, t = t0 + r;

        // softmax #1 (masked)
        float lm = -3.0e38f;
#pragma unroll
        for (int j = 0; j < 32; ++j) lm = fmaxf(lm, raw[rep][j]);
        const float m1 = wave_max32(lm);
        float ls = 0.f;
#pragma unroll
        for (int j = 0; j < 32; ++j) ls += expf(raw[rep][j] - m1);
        const float l1 = wave_sum32(ls);
        const float inv1 = 1.0f / l1;
        const float off = wave_exscan32(ls);

        // cumsum decay + modulated scores
        float lm2 = -3.0e38f;
        float run = 0.f;
#pragma unroll
        for (int j = 0; j < 32; ++j) {
            int s = s0 + j;
            float e = expf(raw[rep][j] - m1);
            run += e;
            float rem = (l1 - (off + run)) * inv1;        // disttot - distcum
            float pos = fabsf((float)(t - s));
            float dsc = sqrtf(fmaxf(rem * pos, 0.f));
            float eff = fminf(fmaxf(expf(dsc * gamma), 1e-5f), 1e5f);
            float mod = (s <= t) ? raw[rep][j] * eff : -3.0e38f;
            raw[rep][j] = mod;
            lm2 = fmaxf(lm2, mod);
        }

        // softmax #2
        const float m2 = wave_max32(lm2);
        float ls2 = 0.f;
#pragma unroll
        for (int j = 0; j < 32; ++j) ls2 += expf(raw[rep][j] - m2);
        const float l2 = wave_sum32(ls2);
        const float inv2 = 1.0f / l2;
#pragma unroll
        for (int j = 0; j < 32; ++j) {
            int s = s0 + j;
            float p = expf(raw[rep][j] - m2) * inv2;
            if (s > t) p = 0.f;
            if (zp && t == 0) p = 0.f;
            u.ph.p[r][s] = (half_t)p;
            u.ph.q[r][s] = (half_t)(p * p);
        }
    }
    __syncthreads();

    // ---- PV: one 16x16 output tile per wave ({mean,cov} x 4 n-tiles) ----
    const int isCov = (wv >= 4), nt = wv & 3;
    const half_t* Vt = (isCov ? vcT : vmT) + ((size_t)by * DKK + nt * 16 + li) * TT;
    const half_t* Abase = isCov ? &u.ph.q[0][0] : &u.ph.p[0][0];

    v8f acc = {};
    const int kbmax = (validN + 31) & ~31;
    for (int kb = 0; kb < kbmax; kb += 32) {
        __builtin_prefetch(Vt + kb + 256, 0, 1);
        v16h A  = load_a16(Abase + (size_t)li * 1024 + kb, hi);
        v16h Bv = load_b16(Vt + kb, hi);
        acc = wmma_f16(A, Bv, acc);
    }
    half_t* outp = isCov ? cch : cmh;
#pragma unroll
    for (int i = 0; i < 8; ++i) {
        int t = t0 + i + 8 * hi;
        outp[((size_t)bb * TT + t) * DD + h * DKK + nt * 16 + li] = (half_t)acc[i];
    }
}

// =====================================================================
// Kernel 3: output projections -> f32 d_out
// =====================================================================
__global__ __launch_bounds__(256) void outproj_kernel(
    const half_t* cmh, const half_t* cch,
    const float* Wo_mean, const float* bo_mean, const float* Wo_cov, const float* bo_cov,
    float* out)
{
    const int bx = blockIdx.x, ny = blockIdx.y, z = blockIdx.z;
    const int m0 = bx * 64, n0 = ny * 64;
    const int tid = threadIdx.x, lane = tid & 31, wv = tid >> 5;
    const int li = lane & 15, hi = (lane >= 16);

    const half_t* X = z ? cch : cmh;
    const float* W = z ? Wo_cov : Wo_mean;
    const float* bias = z ? bo_cov : bo_mean;
    float* O = out + (size_t)z * BT * DD;

    const int t0i = wv, t1i = wv + 8;
    const int tr0 = t0i >> 2, tc0 = t0i & 3, tr1 = t1i >> 2, tc1 = t1i & 3;
    const half_t* ar0 = X + (size_t)(m0 + tr0 * 16 + li) * DD;
    const half_t* ar1 = X + (size_t)(m0 + tr1 * 16 + li) * DD;
    const float* bc0 = W + (size_t)(n0 + tc0 * 16 + li) * DD;
    const float* bc1 = W + (size_t)(n0 + tc1 * 16 + li) * DD;

    v8f acc0 = {}; v8f acc1 = {};
    for (int kb = 0; kb < DD; kb += 32) {
        acc0 = wmma_f16(load_a16(ar0 + kb, hi), load_b32(bc0 + kb, hi), acc0);
        acc1 = wmma_f16(load_a16(ar1 + kb, hi), load_b32(bc1 + kb, hi), acc1);
    }
    const float bn0 = bias[n0 + tc0 * 16 + li];
    const float bn1 = bias[n0 + tc1 * 16 + li];
#pragma unroll
    for (int i = 0; i < 8; ++i) {
        O[(size_t)(m0 + tr0 * 16 + i + 8 * hi) * DD + n0 + tc0 * 16 + li] = acc0[i] + bn0;
        O[(size_t)(m0 + tr1 * 16 + i + 8 * hi) * DD + n0 + tc1 * 16 + li] = acc1[i] + bn1;
    }
}

// =====================================================================
// Host launch
// =====================================================================
extern "C" void kernel_launch(void* const* d_in, const int* in_sizes, int n_in,
                              void* d_out, int out_size, void* d_ws, size_t ws_size,
                              hipStream_t stream)
{
    const float* q_mean  = (const float*)d_in[0];
    const float* q_cov   = (const float*)d_in[1];
    const float* k_mean  = (const float*)d_in[2];
    const float* k_cov   = (const float*)d_in[3];
    const float* v_mean  = (const float*)d_in[4];
    const float* v_cov   = (const float*)d_in[5];
    const float* Wk_mean = (const float*)d_in[6];
    const float* bk_mean = (const float*)d_in[7];
    const float* Wk_cov  = (const float*)d_in[8];
    const float* bk_cov  = (const float*)d_in[9];
    const float* Wv_mean = (const float*)d_in[10];
    const float* bv_mean = (const float*)d_in[11];
    const float* Wv_cov  = (const float*)d_in[12];
    const float* bv_cov  = (const float*)d_in[13];
    const float* Wo_mean = (const float*)d_in[14];
    const float* bo_mean = (const float*)d_in[15];
    const float* Wo_cov  = (const float*)d_in[16];
    const float* bo_cov  = (const float*)d_in[17];
    const float* gammas  = (const float*)d_in[18];
    const int*   zero_pad = (const int*)d_in[20];   // mask (d_in[19]) rebuilt causally

    char* ws = (char*)d_ws;
    size_t off = 0;
    auto take = [&](size_t bytes) -> char* {
        char* p = ws + off;
        off = (off + bytes + 255) & ~(size_t)255;
        return p;
    };
    const size_t NE = (size_t)BT * DD;        // elements per (B,T,D) tensor
    half_t* km_h  = (half_t*)take(NE * 2);
    half_t* sqk_h = (half_t*)take(NE * 2);
    half_t* qm_h  = (half_t*)take(NE * 2);
    half_t* sqq_h = (half_t*)take(NE * 2);
    half_t* vmT   = (half_t*)take(NE * 2);
    half_t* vcT   = (half_t*)take(NE * 2);
    half_t* cmh   = (half_t*)take(NE * 2);
    half_t* cch   = (half_t*)take(NE * 2);
    const size_t NN = (size_t)BB * HH * TT;   // per-(b,h,t) norms
    float* kmsq  = (float*)take(NN * 4);
    float* kcsum = (float*)take(NN * 4);
    float* qmsq  = (float*)take(NN * 4);
    float* qcsum = (float*)take(NN * 4);

    proj_kernel<<<dim3(BT / 64, HH, 6), 256, 0, stream>>>(
        q_mean, q_cov, k_mean, k_cov, v_mean, v_cov,
        Wk_mean, bk_mean, Wk_cov, bk_cov, Wv_mean, bv_mean, Wv_cov, bv_cov,
        km_h, sqk_h, qm_h, sqq_h, vmT, vcT,
        kmsq, kcsum, qmsq, qcsum);

    attn_kernel<<<dim3(TT / 16, BB * HH), 256, 0, stream>>>(
        qm_h, sqq_h, km_h, sqk_h, vmT, vcT,
        qmsq, qcsum, kmsq, kcsum, cmh, cch, gammas, zero_pad);

    outproj_kernel<<<dim3(BT / 64, DD / 64, 2), 256, 0, stream>>>(
        cmh, cch, Wo_mean, bo_mean, Wo_cov, bo_cov, (float*)d_out);
}
